// QNQLSTM_23794118820021
// MI455X (gfx1250) — matrix operand order repaired and checked by reading, not compiled, not run on hardware
//
#include <hip/hip_runtime.h>

// ---------------------------------------------------------------------------
// Problem constants: T=1024, B=64, I=H=512, L=2
//   M = T*B = 65536 rows, K = 512, N = 4H = 2048 gate columns
// ---------------------------------------------------------------------------
#define M_TOT   65536
#define K_DIM   512
#define H_DIM   512
#define N_GATES 2048
#define B_SZ    64
#define TBH     (65536 * 512)
#define BH      (64 * 512)
#define RSTRIDE 1040              // staged B row stride in LDS: 1024B data + 16B pad

typedef __bf16 bf16;
typedef bf16  v16bf __attribute__((ext_vector_type(16)));
typedef float v8f   __attribute__((ext_vector_type(8)));
typedef unsigned int u32;
typedef unsigned int u32x4 __attribute__((ext_vector_type(4)));
typedef unsigned int u32x8 __attribute__((ext_vector_type(8)));

union AFrag { v16bf v; uint4 q[2]; };

// Fast transcendental epilogue: v_exp_f32 + v_rcp_f32 (TRANS ops co-execute
// with the matrix pipe on CDNA5; full-precision division is wasted here).
__device__ __forceinline__ float sigf(float x) {
    return __builtin_amdgcn_rcpf(1.0f + __expf(-x));
}
__device__ __forceinline__ float tanh_fast(float x) {
    float e = __expf(-2.0f * x);
    return (1.0f - e) * __builtin_amdgcn_rcpf(1.0f + e);
}
__device__ __forceinline__ unsigned short f32_to_bf16(float f) {
    unsigned u = __float_as_uint(f);
    return (unsigned short)((u + 0x7FFFu + ((u >> 16) & 1u)) >> 16);
}

// ---------------------------------------------------------------------------
// fp32 -> bf16 conversion (grid-stride)
// ---------------------------------------------------------------------------
__global__ void f32_to_bf16_kernel(const float* __restrict__ in,
                                   unsigned short* __restrict__ out, int n) {
    int i = blockIdx.x * blockDim.x + threadIdx.x;
    int stride = gridDim.x * blockDim.x;
    for (; i < n; i += stride) out[i] = f32_to_bf16(in[i]);
}

// ---------------------------------------------------------------------------
// hhp[b, n] = h0[b,:] . w_hh[n,:] + b_hh[n] + b_ih[n]    (64 x 2048, K=512)
// ---------------------------------------------------------------------------
__global__ __launch_bounds__(256) void hhp_kernel(
    const float* __restrict__ h0l, const float* __restrict__ w_hh,
    const float* __restrict__ b_ih, const float* __restrict__ b_hh,
    float* __restrict__ hhp) {
    int idx = blockIdx.x * 256 + threadIdx.x;
    int n = idx & (N_GATES - 1);
    int b = idx >> 11;
    const float4* hv = (const float4*)(h0l + (size_t)b * K_DIM);
    const float4* wv = (const float4*)(w_hh + (size_t)n * K_DIM);
    float s = 0.0f;
#pragma unroll 8
    for (int k = 0; k < K_DIM / 4; ++k) {
        float4 a = hv[k], w = wv[k];
        s += a.x * w.x + a.y * w.y + a.z * w.z + a.w * w.w;
    }
    hhp[idx] = s + b_ih[n] + b_hh[n];
}

// ---------------------------------------------------------------------------
// Fused GEMM + LSTM epilogue.
// Block: 256 threads (8 waves). Block tile: 256 rows x 32 h-cols x 4 gates.
// TDM stages the block's full B slice once:
//   3D tile X=K(512) x Y=32 rows x Z=4 gate blocks, LDS row stride 1040 B
//   (16B pad per 1024B row rotates banks by 4 per row -> balanced ds_load_b128).
// Wave w owns rows [m0+32w, m0+32w+32): two A fragments, 16 accum tiles.
// Epilogue resolves the LSTM nonlinearity fully in-register (the four gate
// accumulators for one (m,h) sit in the same lane/VGPR slot).
// ---------------------------------------------------------------------------
__global__ __launch_bounds__(256) void lstm_wmma_kernel(
    const unsigned short* __restrict__ A,
    const unsigned short* __restrict__ W,
    const float* __restrict__ hhp,
    const float* __restrict__ c0l,
    unsigned short* __restrict__ hyb,   // bf16 hy out (layer 0) or null
    float* __restrict__ hyf,            // fp32 hy out (layer 1) or null
    float* __restrict__ lasth,
    float* __restrict__ lastc) {
    __shared__ __align__(16) unsigned short Bsh[66560];   // 130 KB, TDM-filled

    const int tid  = threadIdx.x;
    const int wave = tid >> 5;
    const int lane = tid & 31;
    const int hi   = lane >> 4;
    const int j    = lane & 15;

    const int m_wave = blockIdx.x * 256 + wave * 32;
    const int h0     = blockIdx.y * 32;

    // ---- Tensor Data Mover: stage B rows {h0..h0+31} of each gate block ----
    if (tid < 32) {
        unsigned long long ga =
            (unsigned long long)(uintptr_t)(W + (size_t)h0 * K_DIM);
        u32 la = (u32)(uintptr_t)(&Bsh[0]);

        u32x4 g0;
        g0[0] = 1u;                                       // count=1 (valid D#)
        g0[1] = la;                                       // lds_addr
        g0[2] = (u32)ga;                                  // global_addr[31:0]
        g0[3] = ((u32)(ga >> 32) & 0x01FFFFFFu) | (2u << 30); // [56:32] | type=2

        u32x8 g1;
        g1[0] = (1u << 16)     // data_size = 2 bytes
              | (1u << 20)     // pad_enable
              | (7u << 22)     // pad_interval: 256 DW = 1024 B
              | (3u << 25);    // pad_amount: 4 DW = 16 B
        g1[1] = (u32)K_DIM << 16;          // tensor_dim0 low16  @ bits[63:48]
        g1[2] = (u32)N_GATES << 16;        // tensor_dim1 low16  @ bits[95:80]
        g1[3] = (u32)K_DIM << 16;          // tile_dim0          @ bits[127:112]
        g1[4] = 32u | (4u << 16);          // tile_dim1=32, tile_dim2=4
        g1[5] = (u32)K_DIM;                // tensor_dim0_stride low32
        g1[6] = 0u;                        // strd0 hi16, strd1 low16 (262144 -> 0)
        g1[7] = (u32)((K_DIM * H_DIM) >> 16);  // tensor_dim1_stride hi bits (=4)

        u32x4 g2;
        g2[0] = 4u;                        // tensor_dim2 = 4 gate blocks
        g2[1] = 0u; g2[2] = 0u; g2[3] = 0u;
        u32x4 g3 = {0u, 0u, 0u, 0u};

        asm volatile("tensor_load_to_lds %0, %1, %2, %3"
                     :: "s"(g0), "s"(g1), "s"(g2), "s"(g3)
                     : "memory");
    }
    __builtin_amdgcn_s_wait_tensorcnt(0);
    __syncthreads();

    // A fragments: lane holds row (m_wave + j [+16]); hi lanes K offset +8
    const unsigned short* aptr0 = A + (size_t)(m_wave + j) * K_DIM + hi * 8;
    const unsigned short* aptr1 = aptr0 + (size_t)16 * K_DIM;

    // Per-tile LDS byte offsets (tile t = q*2+u -> staged row q*32 + u*16 + j)
    const char* bb = (const char*)Bsh;
    int boff[8];
#pragma unroll
    for (int t = 0; t < 8; ++t)
        boff[t] = ((t >> 1) * 32 + (t & 1) * 16 + j) * RSTRIDE + hi * 16;

    v8f acc0[8] = {};
    v8f acc1[8] = {};

    for (int kk = 0; kk < K_DIM; kk += 32) {
        AFrag a0, a1;
        a0.q[0] = *(const uint4*)(aptr0 + kk);
        a0.q[1] = *(const uint4*)(aptr0 + kk + 16);
        a1.q[0] = *(const uint4*)(aptr1 + kk);
        a1.q[1] = *(const uint4*)(aptr1 + kk + 16);
        if (kk + 32 < K_DIM) {
            __builtin_prefetch(aptr0 + kk + 32, 0, 0);
            __builtin_prefetch(aptr1 + kk + 32, 0, 0);
        }

        AFrag bf[8];
#pragma unroll
        for (int t = 0; t < 8; ++t) {
            const uint4* bp = (const uint4*)(bb + boff[t] + kk * 2);
            bf[t].q[0] = bp[0];
            bf[t].q[1] = bp[2];
        }

#pragma unroll
        for (int t = 0; t < 8; ++t) {
            acc0[t] = __builtin_amdgcn_wmma_f32_16x16x32_bf16(
                false, a0.v, false, bf[t].v, (short)0, acc0[t], false, false);
            acc1[t] = __builtin_amdgcn_wmma_f32_16x16x32_bf16(
                false, a1.v, false, bf[t].v, (short)0, acc1[t], false, false);
        }
    }

    // Fused LSTM epilogue
#pragma unroll
    for (int half = 0; half < 2; ++half) {
        const v8f* acc = half ? acc1 : acc0;
#pragma unroll
        for (int u = 0; u < 2; ++u) {
#pragma unroll
            for (int r = 0; r < 8; ++r) {
                const int m = m_wave + half * 16 + r + hi * 8;
                const int h = h0 + u * 16 + j;
                const int b = m & (B_SZ - 1);
                const float* hb = hhp + (size_t)b * N_GATES + h;

                float ig = acc[0 + u][r] + hb[0 * H_DIM];
                float fg = acc[2 + u][r] + hb[1 * H_DIM];
                float gg = acc[4 + u][r] + hb[2 * H_DIM];
                float og = acc[6 + u][r] + hb[3 * H_DIM];

                float cy = sigf(fg) * c0l[b * H_DIM + h] + sigf(ig) * tanh_fast(gg);
                float hy = sigf(og) * tanh_fast(cy);

                if (hyb) hyb[(size_t)m * H_DIM + h] = f32_to_bf16(hy);
                if (hyf) hyf[(size_t)m * H_DIM + h] = hy;
                if (m >= M_TOT - B_SZ) {
                    lasth[b * H_DIM + h] = hy;
                    lastc[b * H_DIM + h] = cy;
                }
            }
        }
    }
}

// ---------------------------------------------------------------------------
// Launch: convert -> hhp tables -> layer0 fused GEMM -> layer1 fused GEMM
// Workspace (bytes): xb@0 (64MB), hyb0@64MB (64MB), wb0@128MB (2MB),
//                    wb1@130MB (2MB), hhp0@132MB (512KB), hhp1 (+512KB)
// ---------------------------------------------------------------------------
extern "C" void kernel_launch(void* const* d_in, const int* in_sizes, int n_in,
                              void* d_out, int out_size, void* d_ws, size_t ws_size,
                              hipStream_t stream) {
    const float* x      = (const float*)d_in[0];
    const float* h0     = (const float*)d_in[1];
    const float* c0     = (const float*)d_in[2];
    const float* w_ih_0 = (const float*)d_in[3];
    const float* w_hh_0 = (const float*)d_in[4];
    const float* b_ih_0 = (const float*)d_in[5];
    const float* b_hh_0 = (const float*)d_in[6];
    const float* w_ih_1 = (const float*)d_in[7];
    const float* w_hh_1 = (const float*)d_in[8];
    const float* b_ih_1 = (const float*)d_in[9];
    const float* b_hh_1 = (const float*)d_in[10];
    float* out = (float*)d_out;

    char* w = (char*)d_ws;
    unsigned short* xb   = (unsigned short*)(w);
    unsigned short* hyb0 = (unsigned short*)(w + 67108864);
    unsigned short* wb0  = (unsigned short*)(w + 134217728);
    unsigned short* wb1  = (unsigned short*)(w + 136314880);
    float*          hhp0 = (float*)(w + 138412032);
    float*          hhp1 = (float*)(w + 138936320);

    f32_to_bf16_kernel<<<2048, 256, 0, stream>>>(x, xb, M_TOT * K_DIM);
    f32_to_bf16_kernel<<<256, 256, 0, stream>>>(w_ih_0, wb0, N_GATES * K_DIM);
    f32_to_bf16_kernel<<<256, 256, 0, stream>>>(w_ih_1, wb1, N_GATES * K_DIM);

    hhp_kernel<<<(B_SZ * N_GATES) / 256, 256, 0, stream>>>(h0, w_hh_0, b_ih_0, b_hh_0, hhp0);
    hhp_kernel<<<(B_SZ * N_GATES) / 256, 256, 0, stream>>>(h0 + BH, w_hh_1, b_ih_1, b_hh_1, hhp1);

    dim3 grid(M_TOT / 256, H_DIM / 32);   // 256 x 16 blocks
    lstm_wmma_kernel<<<grid, 256, 0, stream>>>(
        xb, wb0, hhp0, c0, hyb0, (float*)nullptr,
        out + TBH, out + TBH + 2 * BH);
    lstm_wmma_kernel<<<grid, 256, 0, stream>>>(
        hyb0, wb1, hhp1, c0 + BH, (unsigned short*)nullptr, out,
        out + TBH + BH, out + TBH + 3 * BH);
}
